// Mamba3Backbone_45681272160521
// MI455X (gfx1250) — matrix operand order repaired and checked by reading, not compiled, not run on hardware
//
#include <hip/hip_runtime.h>
#include <hip/hip_bf16.h>

typedef unsigned short ushort_t;
typedef __attribute__((ext_vector_type(16))) __bf16 bf16x16;
typedef __attribute__((ext_vector_type(8)))  float  f32x8;
typedef int v4i_t __attribute__((__vector_size__(16)));

#define AS_GLOBAL __attribute__((address_space(1)))
#define AS_LDS    __attribute__((address_space(3)))

#if defined(__has_builtin)
#  if __has_builtin(__builtin_amdgcn_global_load_async_to_lds_b128)
#    define HAVE_ASYNC_LDS 1
#  else
#    define HAVE_ASYNC_LDS 0
#  endif
#else
#  define HAVE_ASYNC_LDS 0
#endif

#if HAVE_ASYNC_LDS && __has_builtin(__builtin_amdgcn_s_wait_asynccnt)
#  define WAIT_ASYNC() __builtin_amdgcn_s_wait_asynccnt(0)
#elif HAVE_ASYNC_LDS
#  define WAIT_ASYNC() asm volatile("s_wait_asynccnt 0" ::: "memory")
#else
#  define WAIT_ASYNC() ((void)0)
#endif

// ---- problem constants (from reference) ----
constexpr int Dd    = 1024;
constexpr int DEPTHc= 4;
constexpr int Nn    = 128;   // state dim
constexpr int Pp    = 64;    // head dim
constexpr int Kc    = 4;     // conv width
constexpr int Hh    = 32;    // heads
constexpr int DIc   = 2048;  // EXPAND*D
constexpr int CONVc = DIc + 2*Nn;          // 2304
constexpr int DIPc  = 2*DIc + 2*Nn + Hh;   // 4384
constexpr int FDc   = 2*Dd;  // 2048 (FF hidden)
constexpr int Bc    = 2;
constexpr int Lc    = 1024;
constexpr int BLc   = Bc*Lc; // 2048 rows

// ---- helpers ----
__device__ inline ushort_t f2bf(float f) {
  union { float f; unsigned u; } x; x.f = f;
  unsigned r = x.u + 0x7fffu + ((x.u >> 16) & 1u);
  return (ushort_t)(r >> 16);
}
__device__ inline float siluf(float x) { return x / (1.f + __expf(-x)); }
__device__ inline float geluf(float x) {
  float x3 = x * x * x;
  return 0.5f * x * (1.f + tanhf(0.79788456080286535588f * (x + 0.044715f * x3)));
}
__device__ inline float softplusf(float x) {
  return (x > 20.f) ? x : log1pf(__expf(x));
}

// =====================================================================
// bf16 WMMA GEMM: C[M,N] = A[M,K] * W[K,N]; A,W bf16 row-major, f32 acc.
// Block tile 128x128x32, 8 waves, each wave 64x32 (4x2 WMMA 16x16x32).
// Tiles filled by GLOBAL_LOAD_ASYNC_TO_LDS_B128 (ASYNCcnt-tracked),
// double-buffered, one barrier per K-step, no VGPR staging. Interior
// blocks take a guard-free fast path (block-uniform `edge` test).
// epi==0: store f32 to Cf.  epi==1: store bf16 gelu(x+bias[col]) to Cb.
// =====================================================================
#define BM 128
#define BN 128
#define BK 32

__device__ inline bf16x16 lds_a_frag(const ushort_t* As, int tile, int lane) {
  int m = tile * 16 + (lane & 15);
  const ushort_t* rp = As + m * BK + ((lane >> 4) ? 8 : 0);
  union { uint4 q[2]; bf16x16 v; } u;
  u.q[0] = *(const uint4*)(rp);        // k = k0 .. k0+7
  u.q[1] = *(const uint4*)(rp + 16);   // k = k0+16 .. k0+23
  return u.v;
}
__device__ inline bf16x16 lds_b_frag(const ushort_t* Bs, int tile, int lane) {
  const ushort_t* rp = Bs + tile * 16 + (lane & 15) + ((lane >> 4) ? 8 : 0) * BN; // lane indexes N column
  union { ushort_t e[16]; bf16x16 v; } u;
#pragma unroll
  for (int s = 0; s < 8; ++s) {
    u.e[s]     = rp[s * BN];
    u.e[s + 8] = rp[(s + 16) * BN];
  }
  return u.v;
}

__global__ __launch_bounds__(256)
void gemm_bf16_kernel(const ushort_t* __restrict__ A, const ushort_t* __restrict__ W,
                      float* __restrict__ Cf, ushort_t* __restrict__ Cb,
                      const float* __restrict__ bias,
                      int M, int N, int K, int epi) {
  __shared__ ushort_t As[2][BM * BK];
  __shared__ ushort_t Bs[2][BK * BN];

  const int tid  = threadIdx.x;
  const int lane = tid & 31;
  const int wave = tid >> 5;
  const int wm   = wave & 1;  // 2 waves along M
  const int wn   = wave >> 1; // 4 waves along N
  const int bm   = blockIdx.y;
  const int bn   = blockIdx.x;

  const uint4 zero4 = make_uint4(0u, 0u, 0u, 0u);
  // block-uniform: does this block need bounds handling at all?
  const bool edge = (bm * BM + BM > M) || (bn * BN + BN > N);

  // fill LDS tile `buf` for K-slab `kt`: each thread moves 2x16B of A and
  // 2x16B of B via the async DMA path (no VGPR staging).
  auto tile_load = [&](int kt, int buf) {
    const int k0 = kt * BK;
    if (!edge) {
      // guard-free fast path (wave-uniform branch)
#pragma unroll
      for (int i = 0; i < 2; ++i) {
        int c = tid + i * 256;
        int arow = c >> 2, aoff = (c & 3) * 8;
        int brow = c >> 4, boff = (c & 15) * 8;
#if HAVE_ASYNC_LDS
        __builtin_amdgcn_global_load_async_to_lds_b128(
            (AS_GLOBAL v4i_t*)(A + (size_t)(bm * BM + arow) * K + k0 + aoff),
            (AS_LDS    v4i_t*)&As[buf][arow * BK + aoff], 0, 0);
        __builtin_amdgcn_global_load_async_to_lds_b128(
            (AS_GLOBAL v4i_t*)(W + (size_t)(k0 + brow) * N + bn * BN + boff),
            (AS_LDS    v4i_t*)&Bs[buf][brow * BN + boff], 0, 0);
#else
        uint4 va = *(const uint4*)(A + (size_t)(bm * BM + arow) * K + k0 + aoff);
        uint4 vb = *(const uint4*)(W + (size_t)(k0 + brow) * N + bn * BN + boff);
        *(uint4*)&As[buf][arow * BK + aoff]  = va;
        *(uint4*)&Bs[buf][brow * BN + boff] = vb;
#endif
      }
    } else {
#pragma unroll
      for (int i = 0; i < 2; ++i) {
        int c = tid + i * 256;
        int arow = c >> 2, aoff = (c & 3) * 8;
        int gm = bm * BM + arow;
        int brow = c >> 4, boff = (c & 15) * 8;
        int gn = bn * BN + boff;
#if HAVE_ASYNC_LDS
        if (gm < M)
          __builtin_amdgcn_global_load_async_to_lds_b128(
              (AS_GLOBAL v4i_t*)(A + (size_t)gm * K + k0 + aoff),
              (AS_LDS    v4i_t*)&As[buf][arow * BK + aoff], 0, 0);
        else
          *(uint4*)&As[buf][arow * BK + aoff] = zero4;
        if (gn < N)
          __builtin_amdgcn_global_load_async_to_lds_b128(
              (AS_GLOBAL v4i_t*)(W + (size_t)(k0 + brow) * N + gn),
              (AS_LDS    v4i_t*)&Bs[buf][brow * BN + boff], 0, 0);
        else
          *(uint4*)&Bs[buf][brow * BN + boff] = zero4;
#else
        uint4 va = (gm < M) ? *(const uint4*)(A + (size_t)gm * K + k0 + aoff) : zero4;
        uint4 vb = (gn < N) ? *(const uint4*)(W + (size_t)(k0 + brow) * N + gn) : zero4;
        *(uint4*)&As[buf][arow * BK + aoff]  = va;
        *(uint4*)&Bs[buf][brow * BN + boff] = vb;
#endif
      }
    }
  };

  f32x8 acc[4][2];
  const f32x8 zacc = {0.f, 0.f, 0.f, 0.f, 0.f, 0.f, 0.f, 0.f};
#pragma unroll
  for (int r = 0; r < 4; ++r)
#pragma unroll
    for (int c = 0; c < 2; ++c) acc[r][c] = zacc;

  const int kTiles = K / BK;
  tile_load(0, 0);
  WAIT_ASYNC();
  __syncthreads();

  for (int kt = 0; kt < kTiles; ++kt) {
    const int cur = kt & 1;
    // start filling the other buffer while we compute on `cur`
    if (kt + 1 < kTiles) tile_load(kt + 1, cur ^ 1);
    if (kt + 2 < kTiles) {
      // L2 prefetch for the K-slab after next
      __builtin_prefetch(W + (size_t)((kt + 2) * BK + (tid >> 4)) * N + bn * BN + (tid & 15) * 8, 0, 1);
    }

    bf16x16 af[4], bfv[2];
#pragma unroll
    for (int r = 0; r < 4; ++r) af[r] = lds_a_frag(As[cur], wm * 4 + r, lane);
#pragma unroll
    for (int c = 0; c < 2; ++c) bfv[c] = lds_b_frag(Bs[cur], wn * 2 + c, lane);

#pragma unroll
    for (int r = 0; r < 4; ++r)
#pragma unroll
      for (int c = 0; c < 2; ++c)
        acc[r][c] = __builtin_amdgcn_wmma_f32_16x16x32_bf16(
            false, af[r], false, bfv[c], (short)0, acc[r][c], false, false);

    WAIT_ASYNC();      // next buffer's DMA (this wave's share) complete
    __syncthreads();   // all waves done reading `cur` + all fills visible
  }

  // epilogue + store (C layout: lane<16 -> M=vgpr, N=lane; lane>=16 -> M=8+vgpr)
#pragma unroll
  for (int r = 0; r < 4; ++r) {
#pragma unroll
    for (int c = 0; c < 2; ++c) {
      int row0 = bm * BM + wm * 64 + r * 16 + ((lane >> 4) ? 8 : 0);
      int col  = bn * BN + wn * 32 + c * 16 + (lane & 15);
      if (col < N) {
#pragma unroll
        for (int v = 0; v < 8; ++v) {
          int row = row0 + v;
          if (row < M) {
            float val = acc[r][c][v];
            if (epi == 1) {
              val = geluf(val + (bias ? bias[col] : 0.f));
              Cb[(size_t)row * N + col] = f2bf(val);
            } else {
              Cf[(size_t)row * N + col] = val;
            }
          }
        }
      }
    }
  }
}

// =====================================================================
// elementwise / small kernels
// =====================================================================
__global__ void copy_f32_kernel(float* __restrict__ dst, const float* __restrict__ src, long long n) {
  long long i = (long long)blockIdx.x * blockDim.x + threadIdx.x;
  if (i < n) dst[i] = src[i];
}

__global__ void cvt_bf16_kernel(const float* __restrict__ s, ushort_t* __restrict__ d, long long n) {
  long long i = (long long)blockIdx.x * blockDim.x + threadIdx.x;
  if (i < n) d[i] = f2bf(s[i]);
}

__global__ void silu_kernel(const float* __restrict__ t, float* __restrict__ st, int n) {
  int i = blockIdx.x * blockDim.x + threadIdx.x;
  if (i < n) st[i] = siluf(t[i]);
}

// tiny M=2 projection: out[b, j] = bias[j] + sum_k st[b,k] * W[k, j]
__global__ void adaln_kernel(const float* __restrict__ st, const float* __restrict__ W,
                             const float* __restrict__ b, float* __restrict__ out) {
  int idx = blockIdx.x * blockDim.x + threadIdx.x;
  const int N3 = 3 * Dd;
  if (idx >= Bc * N3) return;
  int j = idx % N3, bb = idx / N3;
  const float* sp = st + bb * Dd;
  float acc = b[j];
  for (int k = 0; k < Dd; ++k) acc += sp[k] * W[(size_t)k * N3 + j];
  out[idx] = acc;
}

// LayerNorm + adaLN modulate -> bf16; optionally also time-flipped copy
__global__ __launch_bounds__(256)
void ln_mod_kernel(const float* __restrict__ x, const float* __restrict__ emb,
                   ushort_t* __restrict__ outf, ushort_t* __restrict__ outb) {
  const int row = blockIdx.x;              // b*L + l
  const int b = row / Lc, l = row % Lc;
  const float* xr = x + (size_t)row * Dd;
  float v[4], s = 0.f, ss = 0.f;
#pragma unroll
  for (int j = 0; j < 4; ++j) {
    v[j] = xr[threadIdx.x + j * 256];
    s += v[j];
    ss += v[j] * v[j];
  }
  __shared__ float rs[256], rss[256];
  rs[threadIdx.x] = s; rss[threadIdx.x] = ss;
  __syncthreads();
  for (int stp = 128; stp > 0; stp >>= 1) {
    if (threadIdx.x < stp) {
      rs[threadIdx.x]  += rs[threadIdx.x + stp];
      rss[threadIdx.x] += rss[threadIdx.x + stp];
    }
    __syncthreads();
  }
  const float mu  = rs[0] * (1.f / Dd);
  const float var = rss[0] * (1.f / Dd) - mu * mu;
  const float inv = rsqrtf(var + 1e-6f);
  const float* eb = emb + b * 3 * Dd;
#pragma unroll
  for (int j = 0; j < 4; ++j) {
    int d = threadIdx.x + j * 256;
    float nv = (v[j] - mu) * inv * (1.f + eb[Dd + d]) + eb[d];
    ushort_t h = f2bf(nv);
    outf[(size_t)row * Dd + d] = h;
    if (outb) outb[((size_t)b * Lc + (Lc - 1 - l)) * Dd + d] = h;
  }
}

// causal depthwise conv (K=4) + SiLU over xBC channels; dt softplus + dA
__global__ void conv_silu_kernel(const float* __restrict__ zx, const float* __restrict__ cw,
                                 const float* __restrict__ cb, const float* __restrict__ dt_bias,
                                 const float* __restrict__ A_log,
                                 float* __restrict__ xh, float* __restrict__ Bm,
                                 float* __restrict__ Cm, float* __restrict__ dtv,
                                 float* __restrict__ dAv) {
  const int CH = CONVc + Hh;
  long long idx = (long long)blockIdx.x * blockDim.x + threadIdx.x;
  if (idx >= (long long)BLc * CH) return;
  int c  = (int)(idx % CH);
  int bl = (int)(idx / CH);
  int l  = bl % Lc, b = bl / Lc;
  if (c < CONVc) {
    float acc = cb[c];
#pragma unroll
    for (int k = 0; k < Kc; ++k) {
      int lk = l - (Kc - 1) + k;
      if (lk >= 0)
        acc += zx[((size_t)b * Lc + lk) * DIPc + DIc + c] * cw[c * Kc + k];
    }
    float sv = siluf(acc);
    size_t base = (size_t)b * Lc + l;
    if (c < DIc)            xh[base * DIc + c] = sv;
    else if (c < DIc + Nn)  Bm[base * Nn + (c - DIc)] = sv;
    else                    Cm[base * Nn + (c - DIc - Nn)] = sv;
  } else {
    int hh = c - CONVc;
    size_t base = (size_t)b * Lc + l;
    float raw = zx[base * DIPc + DIc + CONVc + hh] + dt_bias[hh];
    float dt = softplusf(raw);
    float A = -__expf(A_log[hh]);
    dtv[base * Hh + hh] = dt;
    dAv[base * Hh + hh] = __expf(dt * A);
  }
}

// selective scan: one WG (64 threads) per (b,h); h-state in registers,
// B_t / C_t staged through LDS, y-reduction intra-thread. SiLU(z) gate fused.
__global__ __launch_bounds__(64)
void scan_kernel(const float* __restrict__ xh, const float* __restrict__ Bm,
                 const float* __restrict__ Cm, const float* __restrict__ dtv,
                 const float* __restrict__ dAv, const float* __restrict__ Dp,
                 const float* __restrict__ zx, ushort_t* __restrict__ yg) {
  const int b = blockIdx.x / Hh;
  const int h = blockIdx.x % Hh;
  const int p = threadIdx.x;
  __shared__ float Bsh[Nn], Csh[Nn];
  float hreg[Nn];
#pragma unroll
  for (int n = 0; n < Nn; ++n) hreg[n] = 0.f;
  const float Dph = Dp[h];
  for (int l = 0; l < Lc; ++l) {
    const size_t base = (size_t)b * Lc + l;
    Bsh[p]      = Bm[base * Nn + p];
    Bsh[p + 64] = Bm[base * Nn + p + 64];
    Csh[p]      = Cm[base * Nn + p];
    Csh[p + 64] = Cm[base * Nn + p + 64];
    const float dt_l = dtv[base * Hh + h];
    const float dA_l = dAv[base * Hh + h];
    const float xv   = xh[base * DIc + h * Pp + p];
    const float dbu  = dt_l * xv;
    __syncthreads();
    float y = 0.f;
#pragma unroll
    for (int n = 0; n < Nn; ++n) {
      hreg[n] = hreg[n] * dA_l + dbu * Bsh[n];
      y += hreg[n] * Csh[n];
    }
    y += Dph * xv;
    const float zv = zx[base * DIPc + h * Pp + p];
    yg[base * DIc + h * Pp + p] = f2bf(y * siluf(zv));
    __syncthreads();
  }
}

// x += gate * (fwd + flip(bwd))
__global__ void combine_kernel(float* __restrict__ x, const float* __restrict__ emb,
                               const float* __restrict__ f, const float* __restrict__ bo) {
  long long idx = (long long)blockIdx.x * blockDim.x + threadIdx.x;
  if (idx >= (long long)BLc * Dd) return;
  int d = (int)(idx % Dd);
  int bl = (int)(idx / Dd);
  int l = bl % Lc, b = bl / Lc;
  float gate = emb[b * 3 * Dd + 2 * Dd + d];
  float bv = bo[((size_t)b * Lc + (Lc - 1 - l)) * Dd + d];
  x[idx] += gate * (f[idx] + bv);
}

// x += g2 * (y2 + b2)
__global__ void ff_res_kernel(float* __restrict__ x, const float* __restrict__ ffe,
                              const float* __restrict__ y2, const float* __restrict__ b2) {
  long long idx = (long long)blockIdx.x * blockDim.x + threadIdx.x;
  if (idx >= (long long)BLc * Dd) return;
  int d = (int)(idx % Dd);
  int b = (int)(idx / ((long long)Lc * Dd));
  float g2 = ffe[b * 3 * Dd + 2 * Dd + d];
  x[idx] += g2 * (y2[idx] + b2[d]);
}

// =====================================================================
// host orchestration
// =====================================================================
static inline int cdiv(long long a, long long b) { return (int)((a + b - 1) / b); }

extern "C" void kernel_launch(void* const* d_in, const int* in_sizes, int n_in,
                              void* d_out, int out_size, void* d_ws, size_t ws_size,
                              hipStream_t stream) {
  // input order: x, t, mask, adaln_W, adaln_b, ffada_W, ffada_b,
  // ff_W1, ff_b1, ff_W2, ff_b2,
  // fwd{in_W,conv_w,conv_b,dt_bias,A_log,Dp,out_W}, bwd{...}
  const float* x_in     = (const float*)d_in[0];
  const float* t_in     = (const float*)d_in[1];
  const float* adaln_W  = (const float*)d_in[3];
  const float* adaln_b  = (const float*)d_in[4];
  const float* ffada_W  = (const float*)d_in[5];
  const float* ffada_b  = (const float*)d_in[6];
  const float* ff_W1    = (const float*)d_in[7];
  const float* ff_b1    = (const float*)d_in[8];
  const float* ff_W2    = (const float*)d_in[9];
  const float* ff_b2    = (const float*)d_in[10];
  const float* f_in_W   = (const float*)d_in[11];
  const float* f_conv_w = (const float*)d_in[12];
  const float* f_conv_b = (const float*)d_in[13];
  const float* f_dt_b   = (const float*)d_in[14];
  const float* f_A_log  = (const float*)d_in[15];
  const float* f_Dp     = (const float*)d_in[16];
  const float* f_out_W  = (const float*)d_in[17];
  const float* b_in_W   = (const float*)d_in[18];
  const float* b_conv_w = (const float*)d_in[19];
  const float* b_conv_b = (const float*)d_in[20];
  const float* b_dt_b   = (const float*)d_in[21];
  const float* b_A_log  = (const float*)d_in[22];
  const float* b_Dp     = (const float*)d_in[23];
  const float* b_out_W  = (const float*)d_in[24];

  float* x = (float*)d_out; // residual stream lives in d_out

  size_t off = 0;
  auto wsAlloc = [&](size_t nbytes) -> char* {
    char* p = (char*)d_ws + off;
    off += (nbytes + 255) & ~size_t(255);
    return p;
  };
  // bf16 weight caches
  ushort_t* wb_ff1  = (ushort_t*)wsAlloc((size_t)DEPTHc * Dd * FDc * 2);
  ushort_t* wb_ff2  = (ushort_t*)wsAlloc((size_t)DEPTHc * FDc * Dd * 2);
  ushort_t* wb_inf  = (ushort_t*)wsAlloc((size_t)DEPTHc * Dd * DIPc * 2);
  ushort_t* wb_outf = (ushort_t*)wsAlloc((size_t)DEPTHc * DIc * Dd * 2);
  ushort_t* wb_inb  = (ushort_t*)wsAlloc((size_t)DEPTHc * Dd * DIPc * 2);
  ushort_t* wb_outb = (ushort_t*)wsAlloc((size_t)DEPTHc * DIc * Dd * 2);
  // activations
  float*    st     = (float*)   wsAlloc((size_t)Bc * Dd * 4);
  float*    emb    = (float*)   wsAlloc((size_t)Bc * 3 * Dd * 4);
  float*    ffe    = (float*)   wsAlloc((size_t)Bc * 3 * Dd * 4);
  ushort_t* norm_f = (ushort_t*)wsAlloc((size_t)BLc * Dd * 2);
  ushort_t* norm_b = (ushort_t*)wsAlloc((size_t)BLc * Dd * 2);
  float*    zx     = (float*)   wsAlloc((size_t)BLc * DIPc * 4);
  float*    xh     = (float*)   wsAlloc((size_t)BLc * DIc * 4);
  float*    Bm     = (float*)   wsAlloc((size_t)BLc * Nn * 4);
  float*    Cm     = (float*)   wsAlloc((size_t)BLc * Nn * 4);
  float*    dtv    = (float*)   wsAlloc((size_t)BLc * Hh * 4);
  float*    dAv    = (float*)   wsAlloc((size_t)BLc * Hh * 4);
  ushort_t* yg     = (ushort_t*)wsAlloc((size_t)BLc * DIc * 2);
  float*    f_out  = (float*)   wsAlloc((size_t)BLc * Dd * 4);
  float*    b_out  = (float*)   wsAlloc((size_t)BLc * Dd * 4);
  ushort_t* fn     = (ushort_t*)wsAlloc((size_t)BLc * Dd * 2);
  ushort_t* ffh    = (ushort_t*)wsAlloc((size_t)BLc * FDc * 2);
  float*    ff2o   = (float*)   wsAlloc((size_t)BLc * Dd * 4);
  (void)ws_size; (void)n_in; (void)in_sizes; (void)out_size;

  // init: residual stream, silu(t), one-time bf16 weight conversion
  {
    long long n = (long long)BLc * Dd;
    copy_f32_kernel<<<cdiv(n, 256), 256, 0, stream>>>(x, x_in, n);
    silu_kernel<<<cdiv(Bc * Dd, 256), 256, 0, stream>>>(t_in, st, Bc * Dd);
    long long n1 = (long long)DEPTHc * Dd * FDc;
    long long n2 = (long long)DEPTHc * Dd * DIPc;
    long long n3 = (long long)DEPTHc * DIc * Dd;
    cvt_bf16_kernel<<<cdiv(n1, 256), 256, 0, stream>>>(ff_W1,  wb_ff1,  n1);
    cvt_bf16_kernel<<<cdiv(n1, 256), 256, 0, stream>>>(ff_W2,  wb_ff2,  n1);
    cvt_bf16_kernel<<<cdiv(n2, 256), 256, 0, stream>>>(f_in_W, wb_inf,  n2);
    cvt_bf16_kernel<<<cdiv(n3, 256), 256, 0, stream>>>(f_out_W,wb_outf, n3);
    cvt_bf16_kernel<<<cdiv(n2, 256), 256, 0, stream>>>(b_in_W, wb_inb,  n2);
    cvt_bf16_kernel<<<cdiv(n3, 256), 256, 0, stream>>>(b_out_W,wb_outb, n3);
  }

  const long long nBLD = (long long)BLc * Dd;
  const long long nConv = (long long)BLc * (CONVc + Hh);

  for (int i = 0; i < DEPTHc; ++i) {
    // adaLN embeddings
    adaln_kernel<<<cdiv(Bc * 3 * Dd, 256), 256, 0, stream>>>(
        st, adaln_W + (size_t)i * Dd * 3 * Dd, adaln_b + (size_t)i * 3 * Dd, emb);
    adaln_kernel<<<cdiv(Bc * 3 * Dd, 256), 256, 0, stream>>>(
        st, ffada_W + (size_t)i * Dd * 3 * Dd, ffada_b + (size_t)i * 3 * Dd, ffe);

    // LN + modulate -> bf16 (forward and time-flipped copies)
    ln_mod_kernel<<<BLc, 256, 0, stream>>>(x, emb, norm_f, norm_b);

    for (int dir = 0; dir < 2; ++dir) {
      const ushort_t* nrm  = dir ? norm_b : norm_f;
      const ushort_t* winW = (dir ? wb_inb : wb_inf)  + (size_t)i * Dd * DIPc;
      const ushort_t* woutW= (dir ? wb_outb: wb_outf) + (size_t)i * DIc * Dd;
      const float* cw  = (dir ? b_conv_w : f_conv_w) + (size_t)i * CONVc * Kc;
      const float* cb  = (dir ? b_conv_b : f_conv_b) + (size_t)i * CONVc;
      const float* dtb = (dir ? b_dt_b   : f_dt_b)   + (size_t)i * Hh;
      const float* alg = (dir ? b_A_log  : f_A_log)  + (size_t)i * Hh;
      const float* dpp = (dir ? b_Dp     : f_Dp)     + (size_t)i * Hh;
      float* dst = dir ? b_out : f_out;

      // in-projection GEMM: [2048,1024] x [1024,4384]
      gemm_bf16_kernel<<<dim3(cdiv(DIPc, BN), cdiv(BLc, BM)), 256, 0, stream>>>(
          nrm, winW, zx, nullptr, nullptr, BLc, DIPc, Dd, 0);
      // conv + silu + dt/dA
      conv_silu_kernel<<<cdiv(nConv, 256), 256, 0, stream>>>(
          zx, cw, cb, dtb, alg, xh, Bm, Cm, dtv, dAv);
      // selective scan (fused D-skip + silu(z) gate -> bf16)
      scan_kernel<<<Bc * Hh, 64, 0, stream>>>(xh, Bm, Cm, dtv, dAv, dpp, zx, yg);
      // out-projection GEMM: [2048,2048] x [2048,1024]
      gemm_bf16_kernel<<<dim3(cdiv(Dd, BN), cdiv(BLc, BM)), 256, 0, stream>>>(
          yg, woutW, dst, nullptr, nullptr, BLc, Dd, DIc, 0);
    }

    // residual: x += gate * (fwd + flip(bwd))
    combine_kernel<<<cdiv(nBLD, 256), 256, 0, stream>>>(x, emb, f_out, b_out);

    // FF block
    ln_mod_kernel<<<BLc, 256, 0, stream>>>(x, ffe, fn, nullptr);
    gemm_bf16_kernel<<<dim3(cdiv(FDc, BN), cdiv(BLc, BM)), 256, 0, stream>>>(
        fn, wb_ff1 + (size_t)i * Dd * FDc, nullptr, ffh,
        ff_b1 + (size_t)i * FDc, BLc, FDc, Dd, 1);
    gemm_bf16_kernel<<<dim3(cdiv(Dd, BN), cdiv(BLc, BM)), 256, 0, stream>>>(
        ffh, wb_ff2 + (size_t)i * FDc * Dd, ff2o, nullptr, nullptr, BLc, Dd, FDc, 0);
    ff_res_kernel<<<cdiv(nBLD, 256), 256, 0, stream>>>(x, ffe, ff2o, ff_b2 + (size_t)i * Dd);
  }
}
